// MHSA_63230508531770
// MI455X (gfx1250) — compile-verified, matrix operand order checked by reading, not run on hardware
//
#include <hip/hip_runtime.h>

// ---------------------------------------------------------------------------
// MHSA for MI455X (gfx1250): bf16 WMMA pipeline
//   B=2, L=2048, D=1024, H=16, DH=64
// ---------------------------------------------------------------------------

typedef unsigned short u16;
typedef __attribute__((ext_vector_type(16))) __bf16 v16bf;
typedef __attribute__((ext_vector_type(8)))  float  v8f;

#define BATCH 2
#define SEQ   2048
#define DMODEL 1024
#define NHEAD 16
#define DHEAD 64
#define MROWS (BATCH * SEQ)      // 4096

union FragBF {
    v16bf v;
    uint4 u[2];
    u16   s[16];
};

static __device__ __forceinline__ u16 f2bf(float x) {
    unsigned u = __float_as_uint(x);
    unsigned r = u + 0x7FFFu + ((u >> 16) & 1u);
    return (u16)(r >> 16);
}

static __device__ __forceinline__ v8f wmma_bf16(v16bf a, v16bf b, v8f c) {
    // D = A(16x32) * B(32x16) + C, f32 accumulate
    return __builtin_amdgcn_wmma_f32_16x16x32_bf16(
        /*neg_a=*/false, a, /*neg_b=*/false, b,
        /*c_mod=*/(short)0, c, /*reuse_a=*/false, /*reuse_b=*/false);
}

// Load one 16x32 bf16 fragment (A-layout, or B-layout of a [N,K] row-major
// operand): lane l supplies row (l&15) of the tile; the caller passes
// kb = k0 + (lane>>4)*8 ; we fetch 8 halves at kb and 8 at kb+16.
static __device__ __forceinline__ v16bf load_frag_g(const u16* __restrict__ base,
                                                    int stride, int row, int kb) {
    FragBF f;
    const u16* p = base + (size_t)row * (size_t)stride + kb;
    f.u[0] = *reinterpret_cast<const uint4*>(p);
    f.u[1] = *reinterpret_cast<const uint4*>(p + 16);
    return f.v;
}

static __device__ __forceinline__ float redmax16(float v) {
    v = fmaxf(v, __shfl_xor(v, 1, 32));
    v = fmaxf(v, __shfl_xor(v, 2, 32));
    v = fmaxf(v, __shfl_xor(v, 4, 32));
    v = fmaxf(v, __shfl_xor(v, 8, 32));
    return v;
}
static __device__ __forceinline__ float redsum16(float v) {
    v += __shfl_xor(v, 1, 32);
    v += __shfl_xor(v, 2, 32);
    v += __shfl_xor(v, 4, 32);
    v += __shfl_xor(v, 8, 32);
    return v;
}

// ---------------------------------------------------------------------------
// Stage 1: fp32 -> bf16 (vectorized)
// ---------------------------------------------------------------------------
struct alignas(8) H4 { u16 x, y, z, w; };

__global__ void __launch_bounds__(256)
f32_to_bf16_k(const float* __restrict__ in, u16* __restrict__ out, int n4) {
    int i = blockIdx.x * blockDim.x + threadIdx.x;
    if (i >= n4) return;
    float4 f = reinterpret_cast<const float4*>(in)[i];
    H4 h;
    h.x = f2bf(f.x); h.y = f2bf(f.y); h.z = f2bf(f.z); h.w = f2bf(f.w);
    reinterpret_cast<H4*>(out)[i] = h;
}

// ---------------------------------------------------------------------------
// Stage 2: QKV projection  qkv[4096,3072] = X[4096,1024] @ Wqkv[3072,1024]^T
// Each wave computes a 32x64 tile (2x4 grid of 16x16 WMMA accumulators).
// Epilogue scatters: Q,K -> [B,H,L,DH] bf16 ; V -> transposed [B,H,DH,L] bf16
// ---------------------------------------------------------------------------
__global__ void __launch_bounds__(256)
qkv_gemm_k(const u16* __restrict__ X, const u16* __restrict__ W,
           u16* __restrict__ Qb, u16* __restrict__ Kb, u16* __restrict__ Vt) {
    const int Kdim = DMODEL;          // 1024
    const int TN   = (3 * DMODEL) / 64; // 48 tiles of 64 cols
    int lane = threadIdx.x & 31;
    int wid  = blockIdx.x * 8 + (threadIdx.x >> 5);
    int wm = wid / TN, wn = wid % TN;
    if (wm >= MROWS / 32) return;
    int m0 = wm * 32, n0 = wn * 64;
    int lane15 = lane & 15, half = lane >> 4;

    v8f acc[2][4];
    const v8f vz = {0.f, 0.f, 0.f, 0.f, 0.f, 0.f, 0.f, 0.f};
#pragma unroll
    for (int mt = 0; mt < 2; ++mt)
#pragma unroll
        for (int nt = 0; nt < 4; ++nt) acc[mt][nt] = vz;

    for (int k0 = 0; k0 < Kdim; k0 += 32) {
        int kb = k0 + half * 8;
        v16bf a0 = load_frag_g(X, Kdim, m0 + lane15, kb);
        v16bf a1 = load_frag_g(X, Kdim, m0 + 16 + lane15, kb);
        v16bf b0 = load_frag_g(W, Kdim, n0 +  0 + lane15, kb);
        v16bf b1 = load_frag_g(W, Kdim, n0 + 16 + lane15, kb);
        v16bf b2 = load_frag_g(W, Kdim, n0 + 32 + lane15, kb);
        v16bf b3 = load_frag_g(W, Kdim, n0 + 48 + lane15, kb);
        acc[0][0] = wmma_bf16(a0, b0, acc[0][0]);
        acc[0][1] = wmma_bf16(a0, b1, acc[0][1]);
        acc[0][2] = wmma_bf16(a0, b2, acc[0][2]);
        acc[0][3] = wmma_bf16(a0, b3, acc[0][3]);
        acc[1][0] = wmma_bf16(a1, b0, acc[1][0]);
        acc[1][1] = wmma_bf16(a1, b1, acc[1][1]);
        acc[1][2] = wmma_bf16(a1, b2, acc[1][2]);
        acc[1][3] = wmma_bf16(a1, b3, acc[1][3]);
    }

    // Epilogue: C-layout element (vgpr r, lane) = row r+8*half, col lane15
#pragma unroll
    for (int mt = 0; mt < 2; ++mt) {
        int mrow = m0 + mt * 16 + 8 * half;
#pragma unroll
        for (int nt = 0; nt < 4; ++nt) {
            int nb     = n0 + nt * 16;
            int region = nb >> 10;          // 0=Q 1=K 2=V
            int nn     = nb & 1023;
            int h      = nn >> 6;
            int dh     = (nn & 63) + lane15;
#pragma unroll
            for (int r = 0; r < 8; ++r) {
                int m = mrow + r;
                int b = m >> 11, l = m & 2047;
                u16 val = f2bf(acc[mt][nt][r]);
                size_t bh = (size_t)(b * NHEAD + h);
                if (region == 0)
                    Qb[(bh * SEQ + l) * DHEAD + dh] = val;
                else if (region == 1)
                    Kb[(bh * SEQ + l) * DHEAD + dh] = val;
                else
                    Vt[(bh * DHEAD + dh) * SEQ + l] = val;
            }
        }
    }
}

// ---------------------------------------------------------------------------
// Stage 3: causal flash attention. One wave owns 16 query rows of one (b,h).
// Per 32-key chunk: 4 score WMMAs, online softmax, P staged through LDS
// (C-layout store -> A-fragment ds_load_b128), 4 PV WMMAs vs transposed V.
// ---------------------------------------------------------------------------
#define ATT_WAVES 4
#define PL_STRIDE 40   // halves per LDS row (80B: 16B aligned, bank-skewed)

__global__ void __launch_bounds__(ATT_WAVES * 32)
attn_k(const u16* __restrict__ Qb, const u16* __restrict__ Kb,
       const u16* __restrict__ Vt, u16* __restrict__ AO) {
    __shared__ __align__(16) u16 plds[ATT_WAVES][16 * PL_STRIDE];

    int lane = threadIdx.x & 31;
    int wav  = threadIdx.x >> 5;
    int wid  = blockIdx.x * ATT_WAVES + wav;
    int qt = wid & (SEQ / 16 - 1);       // 0..127 query tile
    int bh = wid >> 7;                   // 0..31  (b*H + h)
    if (bh >= BATCH * NHEAD) return;
    int qb0 = qt * 16;
    int lane15 = lane & 15, half = lane >> 4;

    const u16* qptr = Qb + ((size_t)bh * SEQ + qb0) * DHEAD;
    const u16* kptr = Kb + (size_t)bh * SEQ * DHEAD;
    const u16* vptr = Vt + (size_t)bh * DHEAD * SEQ;

    // Q fragments for both 32-wide dh chunks (reused across whole key loop)
    v16bf qf0 = load_frag_g(qptr, DHEAD, lane15, 0 * 32 + half * 8);
    v16bf qf1 = load_frag_g(qptr, DHEAD, lane15, 1 * 32 + half * 8);

    const v8f vz = {0.f, 0.f, 0.f, 0.f, 0.f, 0.f, 0.f, 0.f};
    v8f o[4] = {vz, vz, vz, vz};
    float m_run[8], l_run[8];
#pragma unroll
    for (int r = 0; r < 8; ++r) { m_run[r] = -1e30f; l_run[r] = 0.f; }

    u16* pl = plds[wav];
    int nch = (qb0 + 47) >> 5;           // 32-key chunks, causally limited

    for (int c = 0; c < nch; ++c) {
        int kbk = c * 32;
        // --- scores: two 16x16 tiles (keys kbk..+15, kbk+16..+31) ---
        v8f s0 = vz, s1 = vz;
        {
            v16bf kf;
            kf = load_frag_g(kptr, DHEAD, kbk +  0 + lane15, 0 * 32 + half * 8);
            s0 = wmma_bf16(qf0, kf, s0);
            kf = load_frag_g(kptr, DHEAD, kbk +  0 + lane15, 1 * 32 + half * 8);
            s0 = wmma_bf16(qf1, kf, s0);
            kf = load_frag_g(kptr, DHEAD, kbk + 16 + lane15, 0 * 32 + half * 8);
            s1 = wmma_bf16(qf0, kf, s1);
            kf = load_frag_g(kptr, DHEAD, kbk + 16 + lane15, 1 * 32 + half * 8);
            s1 = wmma_bf16(qf1, kf, s1);
        }
        // --- scale + causal mask + online softmax ---
        float sa[2][8], pa[2][8];
#pragma unroll
        for (int r = 0; r < 8; ++r) {
            int q = qb0 + r + 8 * half;
            float v0 = s0[r] * 0.125f;              // 1/sqrt(64)
            float v1 = s1[r] * 0.125f;
            if (kbk + lane15 > q)      v0 = -1e30f;
            if (kbk + 16 + lane15 > q) v1 = -1e30f;
            sa[0][r] = v0; sa[1][r] = v1;
        }
#pragma unroll
        for (int r = 0; r < 8; ++r) {
            float rowm = redmax16(fmaxf(sa[0][r], sa[1][r]));
            float mn   = fmaxf(m_run[r], rowm);
            float fr   = __expf(m_run[r] - mn);
            float p0   = __expf(sa[0][r] - mn);
            float p1   = __expf(sa[1][r] - mn);
            float rs   = redsum16(p0 + p1);
            l_run[r]   = l_run[r] * fr + rs;
            m_run[r]   = mn;
            pa[0][r] = p0; pa[1][r] = p1;
            o[0][r] *= fr; o[1][r] *= fr; o[2][r] *= fr; o[3][r] *= fr;
        }
        // --- P (16x32) through LDS: C-layout stores -> A-fragment reload ---
#pragma unroll
        for (int r = 0; r < 8; ++r) {
            int row = r + 8 * half;
            pl[row * PL_STRIDE + lane15]      = f2bf(pa[0][r]);
            pl[row * PL_STRIDE + 16 + lane15] = f2bf(pa[1][r]);
        }
        asm volatile("s_wait_dscnt 0" ::: "memory");   // LDS stores visible
        FragBF pf;
        pf.u[0] = *reinterpret_cast<const uint4*>(pl + lane15 * PL_STRIDE + half * 8);
        pf.u[1] = *reinterpret_cast<const uint4*>(pl + lane15 * PL_STRIDE + half * 8 + 16);
        asm volatile("s_wait_dscnt 0" ::: "memory");   // loads done before reuse
        // --- PV: o[t] += P(16x32) @ V(32keys x 16dh) for 4 dh tiles ---
#pragma unroll
        for (int t = 0; t < 4; ++t) {
            v16bf vf = load_frag_g(vptr, SEQ, t * 16 + lane15, kbk + half * 8);
            o[t] = wmma_bf16(pf.v, vf, o[t]);
        }
    }

    // --- normalize, write bf16 attn output [B*L, D] row-major ---
    int b = bh >> 4, h = bh & 15;
#pragma unroll
    for (int r = 0; r < 8; ++r) {
        float inv = 1.0f / l_run[r];
        size_t row = (size_t)b * SEQ + qb0 + r + 8 * half;
#pragma unroll
        for (int t = 0; t < 4; ++t) {
            int col = h * DHEAD + t * 16 + lane15;
            AO[row * DMODEL + col] = f2bf(o[t][r] * inv);
        }
    }
}

// ---------------------------------------------------------------------------
// Stage 4: output projection  out[4096,1024] = AO[4096,1024] @ Wp[1024,1024]^T
// ---------------------------------------------------------------------------
__global__ void __launch_bounds__(256)
proj_gemm_k(const u16* __restrict__ A, const u16* __restrict__ W,
            float* __restrict__ Out) {
    const int Kdim = DMODEL;
    const int TN   = DMODEL / 64;        // 16
    int lane = threadIdx.x & 31;
    int wid  = blockIdx.x * 8 + (threadIdx.x >> 5);
    int wm = wid / TN, wn = wid % TN;
    if (wm >= MROWS / 32) return;
    int m0 = wm * 32, n0 = wn * 64;
    int lane15 = lane & 15, half = lane >> 4;

    v8f acc[2][4];
    const v8f vz = {0.f, 0.f, 0.f, 0.f, 0.f, 0.f, 0.f, 0.f};
#pragma unroll
    for (int mt = 0; mt < 2; ++mt)
#pragma unroll
        for (int nt = 0; nt < 4; ++nt) acc[mt][nt] = vz;

    for (int k0 = 0; k0 < Kdim; k0 += 32) {
        int kb = k0 + half * 8;
        v16bf a0 = load_frag_g(A, Kdim, m0 + lane15, kb);
        v16bf a1 = load_frag_g(A, Kdim, m0 + 16 + lane15, kb);
        v16bf b0 = load_frag_g(W, Kdim, n0 +  0 + lane15, kb);
        v16bf b1 = load_frag_g(W, Kdim, n0 + 16 + lane15, kb);
        v16bf b2 = load_frag_g(W, Kdim, n0 + 32 + lane15, kb);
        v16bf b3 = load_frag_g(W, Kdim, n0 + 48 + lane15, kb);
        acc[0][0] = wmma_bf16(a0, b0, acc[0][0]);
        acc[0][1] = wmma_bf16(a0, b1, acc[0][1]);
        acc[0][2] = wmma_bf16(a0, b2, acc[0][2]);
        acc[0][3] = wmma_bf16(a0, b3, acc[0][3]);
        acc[1][0] = wmma_bf16(a1, b0, acc[1][0]);
        acc[1][1] = wmma_bf16(a1, b1, acc[1][1]);
        acc[1][2] = wmma_bf16(a1, b2, acc[1][2]);
        acc[1][3] = wmma_bf16(a1, b3, acc[1][3]);
    }
#pragma unroll
    for (int mt = 0; mt < 2; ++mt) {
        int mrow = m0 + mt * 16 + 8 * half;
#pragma unroll
        for (int nt = 0; nt < 4; ++nt) {
            int n = n0 + nt * 16 + lane15;
#pragma unroll
            for (int r = 0; r < 8; ++r)
                Out[(size_t)(mrow + r) * DMODEL + n] = acc[mt][nt][r];
        }
    }
}

// ---------------------------------------------------------------------------
// Host launcher
// ---------------------------------------------------------------------------
extern "C" void kernel_launch(void* const* d_in, const int* in_sizes, int n_in,
                              void* d_out, int out_size, void* d_ws, size_t ws_size,
                              hipStream_t stream) {
    (void)in_sizes; (void)n_in; (void)out_size; (void)ws_size;
    const float* x      = (const float*)d_in[0];
    const float* w_qkv  = (const float*)d_in[1];
    const float* w_proj = (const float*)d_in[2];
    float* out = (float*)d_out;

    char* ws = (char*)d_ws;
    const size_t XB  = (size_t)MROWS * DMODEL;          // 4M elems
    const size_t WQE = (size_t)3 * DMODEL * DMODEL;     // 3M elems
    const size_t WPE = (size_t)DMODEL * DMODEL;         // 1M elems
    const size_t QKVE = (size_t)BATCH * NHEAD * SEQ * DHEAD; // 4M elems each

    u16* xb  = (u16*)(ws);
    u16* wqb = (u16*)(ws + 2 * XB);
    u16* wpb = (u16*)(ws + 2 * XB + 2 * WQE);
    u16* Qb  = (u16*)(ws + 2 * XB + 2 * WQE + 2 * WPE);
    u16* Kb  = (u16*)((char*)Qb + 2 * QKVE);
    u16* Vt  = (u16*)((char*)Kb + 2 * QKVE);
    u16* AO  = (u16*)((char*)Vt + 2 * QKVE);

    // Stage 1: converts
    {
        int n4 = (int)(XB / 4);
        f32_to_bf16_k<<<(n4 + 255) / 256, 256, 0, stream>>>(x, xb, n4);
        n4 = (int)(WQE / 4);
        f32_to_bf16_k<<<(n4 + 255) / 256, 256, 0, stream>>>(w_qkv, wqb, n4);
        n4 = (int)(WPE / 4);
        f32_to_bf16_k<<<(n4 + 255) / 256, 256, 0, stream>>>(w_proj, wpb, n4);
    }
    // Stage 2: QKV GEMM — (4096/32) * (3072/64) = 6144 waves, 8 per block
    qkv_gemm_k<<<6144 / 8, 256, 0, stream>>>(xb, wqb, Qb, Kb, Vt);
    // Stage 3: attention — 32 (b,h) * 128 q-tiles = 4096 waves, 4 per block
    attn_k<<<4096 / ATT_WAVES, ATT_WAVES * 32, 0, stream>>>(Qb, Kb, Vt, AO);
    // Stage 4: projection — (4096/32) * (1024/64) = 2048 waves, 8 per block
    proj_gemm_k<<<2048 / 8, 256, 0, stream>>>(AO, wpb, out);
}
